// PointNetSetAbstraction_66700842107334
// MI455X (gfx1250) — compile-verified, matrix operand order checked by reading, not run on hardware
//
#include <hip/hip_runtime.h>

// ---------------- problem constants ----------------
constexpr int  Bb  = 16;
constexpr int  Nn  = 4096;
constexpr int  Dd  = 64;
constexpr int  NPt = 512;
constexpr int  Kn  = 32;          // neighbors per center
constexpr int  M0c = 64;
constexpr int  M1c = 64;
constexpr int  M2c = 128;
constexpr long LG  = (long)Bb * NPt * Kn;   // 262144 grouped columns
constexpr long LQ  = (long)Bb * NPt;        // 8192 pooled columns
constexpr int  CLp = 160;                    // 131 -> 160 padded concat channels
constexpr float R2 = 0.16f;                  // RAD^2

typedef _Float16 h8   __attribute__((ext_vector_type(8)));
typedef _Float16 v16h __attribute__((ext_vector_type(16)));
typedef float    v8f  __attribute__((ext_vector_type(8)));

// =====================================================================
// WMMA GEMM:  C[M x L] = A[M x Kd] * B[Kd x L] (+bias), strided-batched.
//   A: row-major f16 (Kd contiguous)      == CDNA5 A-fragment layout
//   B: column-major f16 (Kd contiguous)   == CDNA5 B-fragment layout
//   C: column-major f32 (M contiguous)    == CDNA5 C/D tile layout
// Register blocking: each wave owns a 16-column strip and MT row-tiles,
// so each B fragment (the big streaming operand) is loaded once and fed
// to MT WMMAs -> WMMA:VMEM intensity up ~2x vs 1-tile waves.
// M % (16*MT) == 0, Kd % 32 == 0, L % 64 == 0 (guaranteed by padding).
// =====================================================================
template<int MT>
__global__ __launch_bounds__(128)
void wmma_gemm_kernel(const _Float16* __restrict__ A, long lda, long sA,
                      const _Float16* __restrict__ Bm, long ldb, long sB,
                      float* __restrict__ C, long ldc, long sC,
                      const float* __restrict__ bias, int Kd)
{
    const int lane = threadIdx.x & 31;
    const int wave = threadIdx.x >> 5;
    const int tm0  = blockIdx.x * MT;            // first 16-row tile
    const int tn   = blockIdx.y * 4 + wave;      // 16-col tile
    const int bz   = blockIdx.z;
    const int h    = lane >> 4;                  // half-wave select
    const int col  = tn * 16 + (lane & 15);

    const _Float16* Ap = A + (long)bz * sA
                         + (long)(tm0 * 16 + (lane & 15)) * lda + h * 8;
    const _Float16* Bp = Bm + (long)bz * sB + (long)col * ldb + h * 8;

    v8f acc[MT];
    #pragma unroll
    for (int i = 0; i < MT; ++i) acc[i] = {};

    for (int kb = 0; kb < Kd; kb += 32) {
        __builtin_prefetch(Bp + kb + 256, 0, 3);   // near-temporal prefetch
        h8 b0 = *(const h8*)(Bp + kb);
        h8 b1 = *(const h8*)(Bp + kb + 16);
        v16h bw = __builtin_shufflevector(b0, b1,
                     0,1,2,3,4,5,6,7,8,9,10,11,12,13,14,15);
        #pragma unroll
        for (int mt = 0; mt < MT; ++mt) {
            const _Float16* ap = Ap + (long)(mt * 16) * lda + kb;
            h8 a0 = *(const h8*)(ap);
            h8 a1 = *(const h8*)(ap + 16);
            v16h av = __builtin_shufflevector(a0, a1,
                         0,1,2,3,4,5,6,7,8,9,10,11,12,13,14,15);
            acc[mt] = __builtin_amdgcn_wmma_f32_16x16x32_f16(
                          false, av, false, bw, (short)0, acc[mt], false, false);
        }
    }
    #pragma unroll
    for (int mt = 0; mt < MT; ++mt) {
        if (bias) {
            const float* bp = bias + (tm0 + mt) * 16 + h * 8;
            #pragma unroll
            for (int v = 0; v < 8; ++v) acc[mt][v] += bp[v];
        }
        // acc[mt][v] = row ((tm0+mt)*16 + 8h + v), column `col` -> one 32B store
        float* Cp = C + (long)bz * sC + (long)col * ldc + (tm0 + mt) * 16 + h * 8;
        *(v8f*)Cp = acc[mt];
    }
}

// ---------------- FPS: one block per batch ----------------
__global__ __launch_bounds__(256)
void fps_kernel(const float* __restrict__ xyz, int* __restrict__ fps_idx)
{
    const int b = blockIdx.x;
    const float* px = xyz + (long)b * 3 * Nn;
    __shared__ float dist[Nn];
    __shared__ float rv[256];
    __shared__ int   ri[256];
    const int t = threadIdx.x;
    for (int i = t; i < Nn; i += 256) dist[i] = 1e10f;
    __syncthreads();
    int far = 0;
    for (int it = 0; it < NPt; ++it) {
        if (t == 0) fps_idx[b * NPt + it] = far;
        const float cx = px[far], cy = px[Nn + far], cz = px[2 * Nn + far];
        float bestv = -1.f; int besti = 0;
        for (int i = t; i < Nn; i += 256) {
            const float dx = px[i] - cx, dy = px[Nn + i] - cy, dz = px[2 * Nn + i] - cz;
            const float m = fminf(dist[i], dx * dx + dy * dy + dz * dz);
            dist[i] = m;
            if (m > bestv) { bestv = m; besti = i; }
        }
        rv[t] = bestv; ri[t] = besti;
        __syncthreads();
        for (int s = 128; s > 0; s >>= 1) {
            if (t < s && rv[t + s] > rv[t]) { rv[t] = rv[t + s]; ri[t] = ri[t + s]; }
            __syncthreads();
        }
        far = ri[0];
        __syncthreads();
    }
}

// ---------------- ball query: one thread per (b,n) ----------------
__global__ __launch_bounds__(256)
void ball_query_kernel(const float* __restrict__ xyz, const int* __restrict__ fps_idx,
                       int* __restrict__ nbr)
{
    const int b = blockIdx.y;
    const int n = blockIdx.x * blockDim.x + threadIdx.x;
    if (n >= NPt) return;
    const float* px = xyz + (long)b * 3 * Nn;
    const int ci = fps_idx[b * NPt + n];
    const float cx = px[ci], cy = px[Nn + ci], cz = px[2 * Nn + ci];
    int* o = nbr + ((long)b * NPt + n) * Kn;
    int cnt = 0, first = 0;
    for (int j = 0; j < Nn && cnt < Kn; ++j) {
        const float dx = px[j] - cx, dy = px[Nn + j] - cy, dz = px[2 * Nn + j] - cz;
        if (dx * dx + dy * dy + dz * dz <= R2) {
            if (cnt == 0) first = j;
            o[cnt++] = j;
        }
    }
    for (; cnt < Kn; ++cnt) o[cnt] = first;
}

// ---------------- gather grouped features into f16 col-major ----------------
__global__ __launch_bounds__(256)
void gather_feat_kernel(const float* __restrict__ points, const int* __restrict__ fps_idx,
                        const int* __restrict__ nbr, _Float16* __restrict__ X0)
{
    const long tid = (long)blockIdx.x * blockDim.x + threadIdx.x;
    if (tid >= (long)M2c * LG) return;
    const int  c   = (int)(tid & 127);
    const long col = tid >> 7;          // (b*NP + n)*K + k
    const long bn  = col >> 5;
    const int  b   = (int)(bn >> 9);
    const int  j   = nbr[col];
    const float* pb = points + (long)b * Dd * Nn + (long)(c & 63) * Nn;
    float v = pb[j];
    if (c >= 64) v -= pb[fps_idx[bn]];
    X0[col * M2c + c] = (_Float16)v;
}

// ---------------- rel_xyz + zero pad into Hcat rows 128..159 ----------------
__global__ __launch_bounds__(256)
void relxyz_pad_kernel(const float* __restrict__ xyz, const int* __restrict__ fps_idx,
                       const int* __restrict__ nbr, _Float16* __restrict__ Hcat)
{
    const long tid = (long)blockIdx.x * blockDim.x + threadIdx.x;
    if (tid >= 32 * LG) return;
    const int  r   = (int)(tid & 31);
    const long col = tid >> 5;
    float v = 0.f;
    if (r < 3) {
        const long bn = col >> 5;
        const int  b  = (int)(bn >> 9);
        const float* px = xyz + (long)b * 3 * Nn + (long)r * Nn;
        v = px[nbr[col]] - px[fps_idx[bn]];
    }
    Hcat[col * CLp + 128 + r] = (_Float16)v;
}

// ---------------- weight f32 -> f16 (row-major, K padded) ----------------
__global__ __launch_bounds__(256)
void wconvert_kernel(const float* __restrict__ w, _Float16* __restrict__ wp,
                     int M, int Cin, int Kp)
{
    const long tid = (long)blockIdx.x * blockDim.x + threadIdx.x;
    if (tid >= (long)M * Kp) return;
    const int k = (int)(tid % Kp);
    const int m = (int)(tid / Kp);
    wp[(long)m * Kp + k] = (k < Cin) ? (_Float16)w[(long)m * Cin + k] : (_Float16)0.f;
}

// ---------------- per-channel mean / inv-std over L columns ----------------
__global__ __launch_bounds__(256)
void bn_stats_kernel(const float* __restrict__ Y, int M, long L,
                     float* __restrict__ mean, float* __restrict__ rstd)
{
    const int m = blockIdx.x;
    __shared__ float s1[256], s2[256];
    float a = 0.f, b = 0.f;
    for (long i = threadIdx.x; i < L; i += 256) {
        const float v = Y[i * M + m];
        a += v; b += v * v;
    }
    s1[threadIdx.x] = a; s2[threadIdx.x] = b;
    __syncthreads();
    for (int s = 128; s > 0; s >>= 1) {
        if (threadIdx.x < s) { s1[threadIdx.x] += s1[threadIdx.x + s]; s2[threadIdx.x] += s2[threadIdx.x + s]; }
        __syncthreads();
    }
    if (threadIdx.x == 0) {
        const float mu = s1[0] / (float)L;
        const float var = s2[0] / (float)L - mu * mu;
        mean[m] = mu;
        rstd[m] = rsqrtf(var + 1e-5f);
    }
}

// ---------------- BN + ReLU + f16 convert (col-major out, given ld) ----------------
__global__ __launch_bounds__(256)
void bn_apply_kernel(const float* __restrict__ Y, int M, long L,
                     const float* __restrict__ g, const float* __restrict__ be,
                     const float* __restrict__ mean, const float* __restrict__ rstd,
                     _Float16* __restrict__ H, int ldo)
{
    const long tid = (long)blockIdx.x * blockDim.x + threadIdx.x;
    if (tid >= (long)M * L) return;
    const int  m   = (int)(tid % M);
    const long col = tid / M;
    float v = g[m] * (Y[col * M + m] - mean[m]) * rstd[m] + be[m];
    H[col * ldo + m] = (_Float16)fmaxf(v, 0.f);
}

// ---------------- BN + ReLU + max over K, f32 + f16 pooled output ----------------
__global__ __launch_bounds__(256)
void bn_max_kernel(const float* __restrict__ Y,
                   const float* __restrict__ g, const float* __restrict__ be,
                   const float* __restrict__ mean, const float* __restrict__ rstd,
                   float* __restrict__ Xp, _Float16* __restrict__ Xph)
{
    const long tid = (long)blockIdx.x * blockDim.x + threadIdx.x;
    if (tid >= (long)M2c * LQ) return;
    const int  c  = (int)(tid & 127);
    const long oc = tid >> 7;                      // b*NP + n
    const float* yp = Y + oc * Kn * M2c + c;
    const float gg = g[c], bb = be[c], mu = mean[c], rs = rstd[c];
    float mx = 0.f;                                 // ReLU outputs are >= 0
    #pragma unroll 4
    for (int k = 0; k < Kn; ++k) {
        const float v = fmaxf(gg * (yp[(long)k * M2c] - mu) * rs + bb, 0.f);
        mx = fmaxf(mx, v);
    }
    Xp[oc * M2c + c]  = mx;
    Xph[oc * M2c + c] = (_Float16)mx;
}

// ---------------- x0 f32 -> f16 (col-major + transposed row-major) ----------------
__global__ __launch_bounds__(256)
void x0_convert_kernel(const float* __restrict__ x0,
                       _Float16* __restrict__ x0c, _Float16* __restrict__ x0r)
{
    const long tid = (long)blockIdx.x * blockDim.x + threadIdx.x;
    if (tid >= (long)M2c * LQ) return;
    const int  c   = (int)(tid & 127);
    const long col = tid >> 7;
    const float v  = x0[col * M2c + c];
    x0c[col * M2c + c]   = (_Float16)v;
    x0r[(long)c * LQ + col] = (_Float16)v;
}

// ---------------- row softmax of energy (one wave32 per row) ----------------
__global__ __launch_bounds__(256)
void softmax_rows_kernel(float* __restrict__ E)
{
    const int lane = threadIdx.x & 31;
    const int w    = blockIdx.x * 8 + (threadIdx.x >> 5);
    if (w >= Bb * NPt) return;
    const int b = w >> 9, n = w & 511;
    float* e = E + (long)b * NPt * NPt + n;        // stride NPt across m
    float vals[16];
    float mx = -3.0e38f;
    #pragma unroll
    for (int i = 0; i < 16; ++i) { vals[i] = e[(long)(i * 32 + lane) * NPt]; mx = fmaxf(mx, vals[i]); }
    for (int o = 16; o > 0; o >>= 1) mx = fmaxf(mx, __shfl_xor(mx, o, 32));
    float s = 0.f;
    #pragma unroll
    for (int i = 0; i < 16; ++i) { vals[i] = __expf(vals[i] - mx); s += vals[i]; }
    for (int o = 16; o > 0; o >>= 1) s += __shfl_xor(s, o, 32);
    const float inv = 1.f / s;
    #pragma unroll
    for (int i = 0; i < 16; ++i) e[(long)(i * 32 + lane) * NPt] = vals[i] * inv;
}

// ---------------- column normalization -> f16 attn ----------------
__global__ __launch_bounds__(256)
void attn_colnorm_kernel(const float* __restrict__ E, _Float16* __restrict__ attn)
{
    const int lane = threadIdx.x & 31;
    const int w    = blockIdx.x * 8 + (threadIdx.x >> 5);   // b*NP + m
    if (w >= Bb * NPt) return;
    const float* p = E + (long)w * NPt;
    float s = 0.f;
    for (int i = lane; i < NPt; i += 32) s += p[i];
    for (int o = 16; o > 0; o >>= 1) s += __shfl_xor(s, o, 32);
    const float inv = 1.f / (1e-9f + s);
    _Float16* q = attn + (long)w * NPt;
    for (int i = lane; i < NPt; i += 32) q[i] = (_Float16)(p[i] * inv);
}

// ---------------- diff = x - x2 -> f16 col-major ----------------
__global__ __launch_bounds__(256)
void diff_kernel(const float* __restrict__ Xp, const float* __restrict__ x2,
                 _Float16* __restrict__ dh)
{
    const long tid = (long)blockIdx.x * blockDim.x + threadIdx.x;
    if (tid >= (long)M2c * LQ) return;
    dh[tid] = (_Float16)(Xp[tid] - x2[tid]);
}

// ---------------- final: out = x + alpha * relu(bn(yt)) + beta ----------------
__global__ __launch_bounds__(256)
void final_out_kernel(const float* __restrict__ Xp, const float* __restrict__ Yt,
                      const float* __restrict__ gn, const float* __restrict__ gb,
                      const float* __restrict__ mean, const float* __restrict__ rstd,
                      const float* __restrict__ alpha, const float* __restrict__ beta,
                      float* __restrict__ outp)
{
    const long tid = (long)blockIdx.x * blockDim.x + threadIdx.x;
    if (tid >= (long)Bb * M2c * NPt) return;
    const int n = (int)(tid & 511);
    const int c = (int)((tid >> 9) & 127);
    const int b = (int)(tid >> 16);
    const long col = (long)b * NPt + n;
    const float x  = Xp[col * M2c + c];
    const float yt = Yt[col * M2c + c];
    const float xr = fmaxf(gn[c] * (yt - mean[c]) * rstd[c] + gb[c], 0.f);
    outp[tid] = x + alpha[c] * xr + beta[c];
}

// ---------------- new_xyz output: (B,3,NP) ----------------
__global__ __launch_bounds__(256)
void newxyz_kernel(const float* __restrict__ xyz, const int* __restrict__ fps_idx,
                   float* __restrict__ out)
{
    const int tid = blockIdx.x * blockDim.x + threadIdx.x;
    if (tid >= Bb * 3 * NPt) return;
    const int n = tid & 511;
    const int d = (tid >> 9) % 3;
    const int b = tid / (3 * NPt);
    out[tid] = xyz[(long)b * 3 * Nn + (long)d * Nn + fps_idx[b * NPt + n]];
}

// =====================================================================
extern "C" void kernel_launch(void* const* d_in, const int* in_sizes, int n_in,
                              void* d_out, int out_size, void* d_ws, size_t ws_size,
                              hipStream_t stream)
{
    const float* xyz    = (const float*)d_in[0];
    const float* points = (const float*)d_in[1];
    const float* w0 = (const float*)d_in[2];  const float* b0 = (const float*)d_in[3];
    const float* g0 = (const float*)d_in[4];  const float* be0= (const float*)d_in[5];
    const float* w1 = (const float*)d_in[6];  const float* b1 = (const float*)d_in[7];
    const float* g1 = (const float*)d_in[8];  const float* be1= (const float*)d_in[9];
    const float* w2 = (const float*)d_in[10]; const float* b2 = (const float*)d_in[11];
    const float* g2 = (const float*)d_in[12]; const float* be2= (const float*)d_in[13];
    const float* wl = (const float*)d_in[14]; const float* bl = (const float*)d_in[15];
    const float* gl = (const float*)d_in[16]; const float* bel= (const float*)d_in[17];
    const float* wv = (const float*)d_in[18]; const float* bv = (const float*)d_in[19];
    const float* wt = (const float*)d_in[20]; const float* bt = (const float*)d_in[21];
    const float* gn = (const float*)d_in[22]; const float* gb = (const float*)d_in[23];
    const float* alpha = (const float*)d_in[24]; const float* beta = (const float*)d_in[25];

    float* out_xyz = (float*)d_out;                 // 16*3*512
    float* out_pts = (float*)d_out + Bb * 3 * NPt;  // 16*128*512

    // -------- workspace layout --------
    char* base = (char*)d_ws; size_t off = 0;
    auto alloc = [&](size_t bytes) -> void* {
        off = (off + 255) & ~(size_t)255;
        void* p = base + off; off += bytes; return p;
    };
    int*      fpsI = (int*)alloc((size_t)Bb * NPt * 4);
    int*      nbr  = (int*)alloc((size_t)LG * 4);
    _Float16* w0p  = (_Float16*)alloc((size_t)M0c * 128 * 2);
    _Float16* w1p  = (_Float16*)alloc((size_t)M1c * 64  * 2);
    _Float16* w2p  = (_Float16*)alloc((size_t)M2c * 64  * 2);
    _Float16* wlp  = (_Float16*)alloc((size_t)M2c * CLp * 2);
    _Float16* wvp  = (_Float16*)alloc((size_t)M2c * 128 * 2);
    _Float16* wtp  = (_Float16*)alloc((size_t)M2c * 128 * 2);
    _Float16* X0   = (_Float16*)alloc((size_t)M2c * LG * 2);  // 64 MB
    _Float16* H0   = X0;                    // reuse: X0 dead after GEMM0
    _Float16* H1   = X0 + (size_t)M1c * LG; // second half of X0 region
    float*    Y    = (float*)alloc((size_t)M2c * LG * 4);     // 128 MB GEMM scratch
    _Float16* Hcat = (_Float16*)alloc((size_t)CLp * LG * 2);  // 80 MB
    float*    mean = (float*)alloc(128 * 4);
    float*    rstd = (float*)alloc(128 * 4);
    float*    Xp   = (float*)alloc((size_t)M2c * LQ * 4);
    _Float16* Xph  = (_Float16*)alloc((size_t)M2c * LQ * 2);
    float*    x0f  = (float*)alloc((size_t)M2c * LQ * 4);
    _Float16* x0c  = (_Float16*)alloc((size_t)M2c * LQ * 2);
    _Float16* x0r  = (_Float16*)alloc((size_t)M2c * LQ * 2);
    float*    E    = (float*)alloc((size_t)Bb * NPt * NPt * 4);   // 16 MB
    _Float16* attn = (_Float16*)alloc((size_t)Bb * NPt * NPt * 2);
    float*    x2f  = (float*)alloc((size_t)M2c * LQ * 4);
    _Float16* dh   = (_Float16*)alloc((size_t)M2c * LQ * 2);
    float*    Yt   = (float*)alloc((size_t)M2c * LQ * 4);

    auto gemm = [&](const _Float16* A, long lda, long sA,
                    const _Float16* Bm, long ldb, long sB,
                    float* C, long ldc, long sC,
                    const float* bias, int M, int Kd, long L, int batches, int MT) {
        dim3 g((unsigned)(M / (16 * MT)), (unsigned)(L / 64), (unsigned)batches);
        if (MT == 8)
            wmma_gemm_kernel<8><<<g, 128, 0, stream>>>(A, lda, sA, Bm, ldb, sB, C, ldc, sC, bias, Kd);
        else
            wmma_gemm_kernel<4><<<g, 128, 0, stream>>>(A, lda, sA, Bm, ldb, sB, C, ldc, sC, bias, Kd);
    };
    auto blocks = [](long total) { return (unsigned)((total + 255) / 256); };

    // -------- sampling / grouping --------
    fps_kernel<<<Bb, 256, 0, stream>>>(xyz, fpsI);
    ball_query_kernel<<<dim3(NPt / 256, Bb), 256, 0, stream>>>(xyz, fpsI, nbr);
    newxyz_kernel<<<blocks(Bb * 3 * NPt), 256, 0, stream>>>(xyz, fpsI, out_xyz);

    // -------- weight packing --------
    wconvert_kernel<<<blocks((long)M0c * 128), 256, 0, stream>>>(w0, w0p, M0c, 128, 128);
    wconvert_kernel<<<blocks((long)M1c * 64 ), 256, 0, stream>>>(w1, w1p, M1c, 64, 64);
    wconvert_kernel<<<blocks((long)M2c * 64 ), 256, 0, stream>>>(w2, w2p, M2c, 64, 64);
    wconvert_kernel<<<blocks((long)M2c * CLp), 256, 0, stream>>>(wl, wlp, M2c, 131, CLp);
    wconvert_kernel<<<blocks((long)M2c * 128), 256, 0, stream>>>(wv, wvp, M2c, 128, 128);
    wconvert_kernel<<<blocks((long)M2c * 128), 256, 0, stream>>>(wt, wtp, M2c, 128, 128);

    // -------- grouped features --------
    gather_feat_kernel<<<blocks((long)M2c * LG), 256, 0, stream>>>(points, fpsI, nbr, X0);
    relxyz_pad_kernel<<<blocks(32 * LG), 256, 0, stream>>>(xyz, fpsI, nbr, Hcat);

    // -------- conv stack (WMMA GEMM -> BN stats -> BN/ReLU) --------
    gemm(w0p, 128, 0, X0, 128, 0, Y, 64, 0, b0, M0c, 128, LG, 1, 4);
    bn_stats_kernel<<<M0c, 256, 0, stream>>>(Y, M0c, LG, mean, rstd);
    bn_apply_kernel<<<blocks((long)M0c * LG), 256, 0, stream>>>(Y, M0c, LG, g0, be0, mean, rstd, H0, M0c);

    gemm(w1p, 64, 0, H0, 64, 0, Y, 64, 0, b1, M1c, 64, LG, 1, 4);
    bn_stats_kernel<<<M1c, 256, 0, stream>>>(Y, M1c, LG, mean, rstd);
    bn_apply_kernel<<<blocks((long)M1c * LG), 256, 0, stream>>>(Y, M1c, LG, g1, be1, mean, rstd, H1, M1c);

    gemm(w2p, 64, 0, H1, 64, 0, Y, 128, 0, b2, M2c, 64, LG, 1, 8);
    bn_stats_kernel<<<M2c, 256, 0, stream>>>(Y, M2c, LG, mean, rstd);
    bn_apply_kernel<<<blocks((long)M2c * LG), 256, 0, stream>>>(Y, M2c, LG, g2, be2, mean, rstd, Hcat, CLp);

    gemm(wlp, CLp, 0, Hcat, CLp, 0, Y, 128, 0, bl, M2c, CLp, LG, 1, 8);
    bn_stats_kernel<<<M2c, 256, 0, stream>>>(Y, M2c, LG, mean, rstd);
    bn_max_kernel<<<blocks((long)M2c * LQ), 256, 0, stream>>>(Y, gl, bel, mean, rstd, Xp, Xph);

    // -------- attention --------
    gemm(wvp, 128, 0, Xph, 128, 0, x0f, 128, 0, bv, M2c, 128, LQ, 1, 8);
    x0_convert_kernel<<<blocks((long)M2c * LQ), 256, 0, stream>>>(x0f, x0c, x0r);

    // energy[b] = x[b]^T * x0[b]  (A = Xph reinterpreted as 512x128 row-major)
    gemm(Xph, 128, (long)NPt * M2c, x0c, 128, (long)NPt * M2c,
         E, NPt, (long)NPt * NPt, nullptr, NPt, 128, NPt, Bb, 4);
    softmax_rows_kernel<<<blocks((long)Bb * NPt * 32), 256, 0, stream>>>(E);
    attn_colnorm_kernel<<<blocks((long)Bb * NPt * 32), 256, 0, stream>>>(E, attn);

    // x2[b] = x0[b] * attn[b]  (A = x0 transposed copy, per-batch offset 512)
    gemm(x0r, LQ, (long)NPt, attn, NPt, (long)NPt * NPt,
         x2f, 128, (long)NPt * M2c, nullptr, M2c, NPt, NPt, Bb, 8);

    diff_kernel<<<blocks((long)M2c * LQ), 256, 0, stream>>>(Xp, x2f, dh);
    gemm(wtp, 128, 0, dh, 128, 0, Yt, 128, 0, bt, M2c, 128, LQ, 1, 8);
    bn_stats_kernel<<<M2c, 256, 0, stream>>>(Yt, M2c, LQ, mean, rstd);

    final_out_kernel<<<blocks((long)Bb * M2c * NPt), 256, 0, stream>>>(
        Xp, Yt, gn, gb, mean, rstd, alpha, beta, out_pts);

    (void)in_sizes; (void)n_in; (void)out_size; (void)ws_size;
}
